// LittleBitParallelLinear_59115929862477
// MI455X (gfx1250) — compile-verified, matrix operand order checked by reading, not run on hardware
//
#include <hip/hip_runtime.h>
#include <hip/hip_bf16.h>
#include <stdint.h>

// ---- types -----------------------------------------------------------------
typedef __attribute__((ext_vector_type(16))) _Float16 v16h;
typedef __attribute__((ext_vector_type(8)))  _Float16 v8h;
typedef __attribute__((ext_vector_type(4)))  _Float16 v4h;
typedef __attribute__((ext_vector_type(8)))  float    v8f;
typedef __attribute__((ext_vector_type(4)))  float    f32x4;
typedef __attribute__((ext_vector_type(4)))  unsigned int u32x4;
typedef __attribute__((ext_vector_type(4)))  int      i32x4;

// address-space-qualified pointers for the async-DMA builtin (V4i*1, V4i*3)
typedef __attribute__((address_space(1))) i32x4* gas_i32x4_ptr;
typedef __attribute__((address_space(3))) i32x4* las_i32x4_ptr;

// ---- problem dims ----------------------------------------------------------
#define TOKENS  8192
#define IN_DIM  4096
#define OUT_DIM 4096
#define RANK    2048

// ---- tiling ----------------------------------------------------------------
#define BM 128
#define BN 128
#define BK 32
#define LDS_STRIDE 40   // 32 halves + 8 pad -> conflict-free b128 frag loads

// ---- CDNA5 async global->LDS copy (ASYNCcnt-tracked DMA, no VGPR staging) --
#if defined(__has_builtin)
#if __has_builtin(__builtin_amdgcn_global_load_async_to_lds_b128)
#define HAVE_ASYNC_LDS 1
#endif
#endif
#ifndef HAVE_ASYNC_LDS
#define HAVE_ASYNC_LDS 0
#endif

__device__ __forceinline__ void async_copy_b128(const void* g, void* l) {
#if HAVE_ASYNC_LDS
    __builtin_amdgcn_global_load_async_to_lds_b128(
        (gas_i32x4_ptr)(uintptr_t)g,
        (las_i32x4_ptr)(uintptr_t)l,
        0, 0);
#else
    *(u32x4*)l = *(const u32x4*)g;   // sync fallback
#endif
}

__device__ __forceinline__ void wait_async0() {
#if HAVE_ASYNC_LDS
#if __has_builtin(__builtin_amdgcn_s_wait_asynccnt)
    __builtin_amdgcn_s_wait_asynccnt(0);
#else
    asm volatile("s_wait_asynccnt 0x0" ::: "memory");
#endif
#endif
}

// Fragment gather per CDNA5 16-bit A/B layout (16x32, MxK):
//  lanes 0-15 : row = lane,    halves cover K {0..7, 16..23}
//  lanes 16-31: row = lane-16, halves cover K {8..15, 24..31}
__device__ __forceinline__ v16h load_frag(const _Float16* lds, int rowBase, int lane) {
    const int lh = lane >> 4;
    const int lr = lane & 15;
    const _Float16* p = lds + (rowBase + lr) * LDS_STRIDE + lh * 8;
    v8h lo = *(const v8h*)(p);
    v8h hi = *(const v8h*)(p + 16);
    return __builtin_shufflevector(lo, hi, 0,1,2,3,4,5,6,7,8,9,10,11,12,13,14,15);
}

// ---- prep: u -> sign(u) as f16 [OUT][RANK] ---------------------------------
__global__ __launch_bounds__(256)
void binarize_rows(const float* __restrict__ src, _Float16* __restrict__ dst) {
    const size_t i = ((size_t)blockIdx.x * 256 + threadIdx.x) * 4;
    f32x4 vsrc = *(const f32x4*)(src + i);
    v4h o;
    o.x = (vsrc.x >= 0.f) ? (_Float16)1.0f : (_Float16)-1.0f;
    o.y = (vsrc.y >= 0.f) ? (_Float16)1.0f : (_Float16)-1.0f;
    o.z = (vsrc.z >= 0.f) ? (_Float16)1.0f : (_Float16)-1.0f;
    o.w = (vsrc.w >= 0.f) ? (_Float16)1.0f : (_Float16)-1.0f;
    *(v4h*)(dst + i) = o;
}

// ---- prep: v [IN][RANK] -> sign(v)^T as f16 [RANK][IN] (LDS transpose) -----
__global__ __launch_bounds__(256)
void binarize_transpose(const float* __restrict__ v, _Float16* __restrict__ bvT) {
    __shared__ _Float16 tile[64][72];
    const int k0 = blockIdx.y * 64;             // IN dim
    const int r0 = blockIdx.x * 64;             // RANK dim
    #pragma unroll
    for (int i = 0; i < 16; ++i) {
        int q  = i * 256 + threadIdx.x;
        int kr = q >> 6, rc = q & 63;
        float val = v[(size_t)(k0 + kr) * RANK + r0 + rc];
        tile[kr][rc] = (val >= 0.f) ? (_Float16)1.0f : (_Float16)-1.0f;
    }
    __syncthreads();
    #pragma unroll
    for (int i = 0; i < 16; ++i) {
        int q  = i * 256 + threadIdx.x;
        int rr = q >> 6, kc = q & 63;
        bvT[(size_t)(r0 + rr) * IN_DIM + k0 + kc] = tile[kc][rr];
    }
}

// ---- GEMM1: t[m][r] = (sum_k x[m][k]*h_in[k]*bvT[r][k]) * s[r], f16 out ----
__global__ __launch_bounds__(256)
void gemm1_xhv(const float* __restrict__ x, const _Float16* __restrict__ bvT,
               const float* __restrict__ h_in, const float* __restrict__ s,
               _Float16* __restrict__ t_out) {
    __shared__ _Float16 ldsA[2][BM * LDS_STRIDE];
    __shared__ _Float16 ldsB[2][BN * LDS_STRIDE];

    const int tid  = threadIdx.x;
    const int lane = tid & 31;
    const int wave = tid >> 5;
    const int wm   = wave & 1;
    const int wn   = wave >> 1;
    const int m0   = blockIdx.y * BM;
    const int n0   = blockIdx.x * BN;

    v8f acc[4][2];
    #pragma unroll
    for (int i = 0; i < 4; ++i)
        #pragma unroll
        for (int j = 0; j < 2; ++j) acc[i][j] = (v8f)0.0f;

    f32x4 aReg[4]; f32x4 hReg[4];

    // B tile: pure f16 -> async DMA straight into LDS (no VGPR staging)
    auto asyncLoadB = [&](int buf, int kc) {
        const int k0 = kc * BK;
        #pragma unroll
        for (int i = 0; i < 2; ++i) {
            int q = i * 256 + tid;
            int row = q >> 2, cv = q & 3;
            async_copy_b128(bvT + (size_t)(n0 + row) * IN_DIM + k0 + cv * 8,
                            &ldsB[buf][row * LDS_STRIDE + cv * 8]);
        }
    };
    // A tile: f32 -> fuse *h_in and cvt to f16 through VGPRs
    auto aLoad = [&](int kc) {
        const int k0 = kc * BK;
        #pragma unroll
        for (int i = 0; i < 4; ++i) {
            int q = i * 256 + tid;
            int row = q >> 3, cv = q & 7;
            aReg[i] = *(const f32x4*)(x + (size_t)(m0 + row) * IN_DIM + k0 + cv * 4);
            hReg[i] = *(const f32x4*)(h_in + k0 + cv * 4);
        }
    };
    auto aStore = [&](int buf) {
        #pragma unroll
        for (int i = 0; i < 4; ++i) {
            int q = i * 256 + tid;
            int row = q >> 3, cv = q & 7;
            v4h pk;
            pk.x = (_Float16)(aReg[i].x * hReg[i].x);
            pk.y = (_Float16)(aReg[i].y * hReg[i].y);
            pk.z = (_Float16)(aReg[i].z * hReg[i].z);
            pk.w = (_Float16)(aReg[i].w * hReg[i].w);
            *(v4h*)(&ldsA[buf][row * LDS_STRIDE + cv * 4]) = pk;
        }
    };
    auto compute = [&](int buf) {
        const _Float16* la = ldsA[buf];
        const _Float16* lb = ldsB[buf];
        v16h afr[4], bfr[2];
        #pragma unroll
        for (int tm = 0; tm < 4; ++tm) afr[tm] = load_frag(la, wm * 64 + tm * 16, lane);
        #pragma unroll
        for (int tn = 0; tn < 2; ++tn) bfr[tn] = load_frag(lb, wn * 32 + tn * 16, lane);
        #pragma unroll
        for (int tm = 0; tm < 4; ++tm)
            #pragma unroll
            for (int tn = 0; tn < 2; ++tn)
                acc[tm][tn] = __builtin_amdgcn_wmma_f32_16x16x32_f16(
                    false, afr[tm], false, bfr[tn], (short)0, acc[tm][tn], false, false);
    };

    const int KC = IN_DIM / BK;
    asyncLoadB(0, 0);
    aLoad(0);
    aStore(0);
    wait_async0();
    __syncthreads();
    for (int kc = 0; kc < KC; ++kc) {
        const int cur = kc & 1;
        if (kc + 1 < KC) { asyncLoadB(cur ^ 1, kc + 1); aLoad(kc + 1); }
        compute(cur);
        if (kc + 1 < KC) { aStore(cur ^ 1); wait_async0(); }
        __syncthreads();
    }

    const int lh = lane >> 4, lr = lane & 15;
    #pragma unroll
    for (int tn = 0; tn < 2; ++tn) {
        const int col = n0 + wn * 32 + tn * 16 + lr;
        const float sc = s[col];
        #pragma unroll
        for (int tm = 0; tm < 4; ++tm) {
            const int mbase = m0 + wm * 64 + tm * 16 + lh * 8;  // M = j + 8*(lane>=16)
            #pragma unroll
            for (int j = 0; j < 8; ++j)
                t_out[(size_t)(mbase + j) * RANK + col] = (_Float16)(acc[tm][tn][j] * sc);
        }
    }
}

// ---- GEMM2: y[m][o] = (sum_r t[m][r]*buh[o][r]) * h_out[o], f32 out --------
__global__ __launch_bounds__(256)
void gemm2_tby(const _Float16* __restrict__ t_in, const _Float16* __restrict__ buh,
               const float* __restrict__ h_out, float* __restrict__ y) {
    __shared__ _Float16 ldsA[2][BM * LDS_STRIDE];
    __shared__ _Float16 ldsB[2][BN * LDS_STRIDE];

    const int tid  = threadIdx.x;
    const int lane = tid & 31;
    const int wave = tid >> 5;
    const int wm   = wave & 1;
    const int wn   = wave >> 1;
    const int m0   = blockIdx.y * BM;
    const int n0   = blockIdx.x * BN;

    v8f acc[4][2];
    #pragma unroll
    for (int i = 0; i < 4; ++i)
        #pragma unroll
        for (int j = 0; j < 2; ++j) acc[i][j] = (v8f)0.0f;

    // Both tiles are f16: pure async DMA into LDS, zero VGPR staging.
    auto asyncLoad = [&](int buf, int kc) {
        const int k0 = kc * BK;
        #pragma unroll
        for (int i = 0; i < 2; ++i) {
            int q = i * 256 + tid;
            int row = q >> 2, cv = q & 3;
            async_copy_b128(t_in + (size_t)(m0 + row) * RANK + k0 + cv * 8,
                            &ldsA[buf][row * LDS_STRIDE + cv * 8]);
            async_copy_b128(buh  + (size_t)(n0 + row) * RANK + k0 + cv * 8,
                            &ldsB[buf][row * LDS_STRIDE + cv * 8]);
        }
    };
    auto compute = [&](int buf) {
        const _Float16* la = ldsA[buf];
        const _Float16* lb = ldsB[buf];
        v16h afr[4], bfr[2];
        #pragma unroll
        for (int tm = 0; tm < 4; ++tm) afr[tm] = load_frag(la, wm * 64 + tm * 16, lane);
        #pragma unroll
        for (int tn = 0; tn < 2; ++tn) bfr[tn] = load_frag(lb, wn * 32 + tn * 16, lane);
        #pragma unroll
        for (int tm = 0; tm < 4; ++tm)
            #pragma unroll
            for (int tn = 0; tn < 2; ++tn)
                acc[tm][tn] = __builtin_amdgcn_wmma_f32_16x16x32_f16(
                    false, afr[tm], false, bfr[tn], (short)0, acc[tm][tn], false, false);
    };

    const int KC = RANK / BK;
    asyncLoad(0, 0);
    wait_async0();
    __syncthreads();
    for (int kc = 0; kc < KC; ++kc) {
        const int cur = kc & 1;
        if (kc + 1 < KC) asyncLoad(cur ^ 1, kc + 1);
        compute(cur);
        if (kc + 1 < KC) wait_async0();
        __syncthreads();
    }

    const int lh = lane >> 4, lr = lane & 15;
    #pragma unroll
    for (int tn = 0; tn < 2; ++tn) {
        const int col = n0 + wn * 32 + tn * 16 + lr;
        const float sc = h_out[col];
        #pragma unroll
        for (int tm = 0; tm < 4; ++tm) {
            const int mbase = m0 + wm * 64 + tm * 16 + lh * 8;
            #pragma unroll
            for (int j = 0; j < 8; ++j)
                y[(size_t)(mbase + j) * OUT_DIM + col] = acc[tm][tn][j] * sc;
        }
    }
}

// ---- launch ----------------------------------------------------------------
extern "C" void kernel_launch(void* const* d_in, const int* in_sizes, int n_in,
                              void* d_out, int out_size, void* d_ws, size_t ws_size,
                              hipStream_t stream) {
    (void)in_sizes; (void)n_in; (void)out_size; (void)ws_size;
    const float* x     = (const float*)d_in[0];   // [TOKENS][IN]
    const float* u     = (const float*)d_in[1];   // [OUT][RANK]
    const float* v     = (const float*)d_in[2];   // [IN][RANK]
    const float* s     = (const float*)d_in[3];   // [RANK]
    const float* h_in  = (const float*)d_in[4];   // [IN]
    const float* h_out = (const float*)d_in[5];   // [OUT]
    float* y = (float*)d_out;                     // [TOKENS][OUT]

    // workspace layout (64 MiB total):
    //   buh  f16 [OUT][RANK]    16 MiB
    //   bvT  f16 [RANK][IN]     16 MiB
    //   t    f16 [TOKENS][RANK] 32 MiB
    char* ws = (char*)d_ws;
    _Float16* buh  = (_Float16*)ws;
    _Float16* bvT  = (_Float16*)(ws + (size_t)OUT_DIM * RANK * 2);
    _Float16* tmid = (_Float16*)(ws + (size_t)OUT_DIM * RANK * 2
                                    + (size_t)RANK * IN_DIM * 2);

    binarize_rows<<<(OUT_DIM * RANK) / 1024, 256, 0, stream>>>(u, buh);
    binarize_transpose<<<dim3(RANK / 64, IN_DIM / 64), 256, 0, stream>>>(v, bvT);
    gemm1_xhv<<<dim3(RANK / BN,    TOKENS / BM), 256, 0, stream>>>(x, bvT, h_in, s, tmid);
    gemm2_tby<<<dim3(OUT_DIM / BN, TOKENS / BM), 256, 0, stream>>>(tmid, buh, h_out, y);
}
